// WaveNet_29807073034880
// MI455X (gfx1250) — compile-verified, matrix operand order checked by reading, not run on hardware
//
#include <hip/hip_runtime.h>

// ---------------------------------------------------------------------------
// WaveNet vocoder forward on gfx1250 (MI455X), f16 WMMA (16x16x32, f32 acc).
// Time-major activations [T][Cpad]: GEMMs are out[t][oc] = sum_c X[t][c]W[c][oc]
//   A operand = activations (M=time, K=channel)  -> two contiguous b128/lane
//   B operand = repacked f16 weights (K=channel, N=out-channel)
// Residual-block input tiles staged to LDS via GLOBAL_LOAD_ASYNC_TO_LDS_B128
// (ASYNCcnt) when the toolchain exposes the builtin.
// ---------------------------------------------------------------------------

typedef __attribute__((ext_vector_type(16))) _Float16 v16h;
typedef __attribute__((ext_vector_type(8)))  _Float16 v8h;
typedef __attribute__((ext_vector_type(8)))  float    v8f;
typedef int v4i_vs __attribute__((vector_size(16)));   // matches builtin param

#define RC   120
#define SC   240
#define NQ   256
#define NB   16
#define NMEL 80

#define AS1 __attribute__((address_space(1)))
#define AS3 __attribute__((address_space(3)))

// ---- WMMA layout helpers ---------------------------------------------------
// A (16x32 f16): lane (l=lane&15, hi=lane>>4) holds M=l; elem j:
//     K = (j<8) ? hi*8+j : 16+hi*8+(j-8)   (two runs of 8 consecutive K)
// B (32x16 f16): assumed symmetric: lane holds N=l, same K map.
// C/D (16x16 f32): elem r: M = r + 8*hi, N = lane&15.
__device__ __forceinline__ int kmap(int j, int hi) {
  return (j < 8) ? (hi * 8 + j) : (16 + hi * 8 + (j - 8));
}

__device__ __forceinline__ v8f wmma_f16(v16h a, v16h b, v8f c) {
  return __builtin_amdgcn_wmma_f32_16x16x32_f16(
      false, a, false, b, (short)0, c, false, false);
}

// A operand from a time-major activation array (LDS or global, generic ptr):
// rows t0+l15, channels kt*32 + hi*8 (+0..7) and +16 (+0..7).
__device__ __forceinline__ v16h ldAct(const _Float16* base, int rstride,
                                      int trow, int kt, int lane) {
  const int l15 = lane & 15, hi = lane >> 4;
  const _Float16* p = base + (size_t)(trow + l15) * rstride + kt * 32 + hi * 8;
  v8h lo = *(const v8h*)p;
  v8h up = *(const v8h*)(p + 16);
  return __builtin_shufflevector(lo, up, 0,1,2,3,4,5,6,7,8,9,10,11,12,13,14,15);
}

// B operand: packed weight tile (global), 32B per lane
__device__ __forceinline__ v16h ldPW(const _Float16* __restrict__ g, int tile, int lane) {
  return reinterpret_cast<const v16h*>(g)[tile * 32 + lane];
}

// ---- async 16B global -> LDS copy with fallback ---------------------------
__device__ __forceinline__ void async_copy16(const _Float16* g, _Float16* l) {
#if __has_builtin(__builtin_amdgcn_global_load_async_to_lds_b128)
  __builtin_amdgcn_global_load_async_to_lds_b128(
      (AS1 v4i_vs*)(v4i_vs*)(void*)const_cast<_Float16*>(g),
      (AS3 v4i_vs*)(v4i_vs*)(void*)l, 0, 0);
#else
  *(uint4*)l = *(const uint4*)g;
#endif
}
__device__ __forceinline__ void async_join() {
#if __has_builtin(__builtin_amdgcn_global_load_async_to_lds_b128)
#if __has_builtin(__builtin_amdgcn_s_wait_asynccnt)
  __builtin_amdgcn_s_wait_asynccnt(0);
#else
  asm volatile("s_wait_asynccnt 0x0" ::: "memory");
#endif
#endif
}

// ---------------------------------------------------------------------------
// Weight repack: f32 [M][K] (strided) -> f16 packed tile layout (used as B).
// mode 0: rows identity, valid m < Morig.
// mode 1: 240->256 split-pad (rows 0..119 -> 0..119, 120..239 -> 128..247)
// ---------------------------------------------------------------------------
__global__ void pack_weights(const float* __restrict__ src, _Float16* __restrict__ dst,
                             int nblk, int MT, int KT, int Morig, int Korig, int mode,
                             int bstride, int rstride, int kstride, int koff) {
  int total = nblk * MT * KT * 512;
  int idx = blockIdx.x * blockDim.x + threadIdx.x;
  if (idx >= total) return;
  int per  = MT * KT * 512;
  int b    = idx / per;
  int r    = idx - b * per;
  int tile = r >> 9;
  int lane = (r >> 4) & 31;
  int j    = r & 15;
  int kt   = tile % KT;
  int mt   = tile / KT;
  int hi   = lane >> 4;
  int m    = mt * 16 + (lane & 15);
  int kk   = kt * 32 + kmap(j, hi);
  int om; bool ok;
  if (mode == 1) {
    if (m < 128) { om = m;     ok = (m < 120); }
    else         { om = m - 8; ok = (om < 240); }
  } else {
    om = m; ok = (m < Morig);
  }
  ok = ok && (kk < Korig);
  float v = ok ? src[(long)b * bstride + (long)om * rstride + (long)kk * kstride + koff] : 0.0f;
  dst[idx] = (_Float16)v;
}

// ---------------------------------------------------------------------------
// Upsample (transposed conv, lhs_dilation=256, pad -1): cond[Tpad][96] f16
// cond[t][o] = up_b[o] + sum_{k=k0+256j<800} sum_c up_w[o][c][k]*mel[c][(t+k+1)/256]
// ---------------------------------------------------------------------------
__global__ void upsample_cond(const float* __restrict__ mel, const float* __restrict__ up_w,
                              const float* __restrict__ up_b, _Float16* __restrict__ cond,
                              int Tpad) {
  long idx = (long)blockIdx.x * blockDim.x + threadIdx.x;
  if (idx >= (long)96 * Tpad) return;
  int c = (int)(idx % 96);
  int t = (int)(idx / 96);
  if (c >= NMEL) { cond[(long)t * 96 + c] = (_Float16)0.0f; return; }
  float acc = up_b[c];
  int k0 = 255 - (t & 255);
  #pragma unroll
  for (int jj = 0; jj < 4; ++jj) {
    int k = k0 + 256 * jj;
    if (k < 800) {
      int f = (t + k + 1) >> 8;
      if (f < 160) {
        const float* wr = up_w + (long)c * 80 * 800 + k;
        const float* mr = mel + f;
        for (int ci = 0; ci < NMEL; ++ci)
          acc += wr[ci * 800] * mr[ci * 160];
      }
    }
  }
  cond[(long)t * 96 + c] = (_Float16)acc;
}

// res0[t][c] = in_w[c]*wav[t] + in_b[c]; channels 120..127 zero
__global__ void input_conv(const float* __restrict__ wav, const float* __restrict__ in_w,
                           const float* __restrict__ in_b, _Float16* __restrict__ res,
                           int Tpad) {
  long idx = (long)blockIdx.x * blockDim.x + threadIdx.x;
  if (idx >= (long)128 * Tpad) return;
  int c = (int)(idx & 127);
  int t = (int)(idx >> 7);
  float v = (c < RC) ? (in_w[c] * wav[t] + in_b[c]) : 0.0f;
  res[idx] = (_Float16)v;
}

// ---------------------------------------------------------------------------
// One residual block. 64 time rows per workgroup, 256 threads (8 waves).
//   z[t][0:256p] = res[t]@Wc1 + res[t-d]@Wc0 + cond[t]@Wcond + bias
//   h[t][c] = tanh(z[t][c]) * sigmoid(z[t][128+c])      c in 0..127(pad)
//   skip[t] += h[t]@Wskip ; res_out[t] = res_in[t] + h[t]@Wres
// ---------------------------------------------------------------------------
__global__ void __launch_bounds__(256)
wavenet_block(const _Float16* __restrict__ res_in, _Float16* __restrict__ res_out,
              const _Float16* __restrict__ cond,
              const _Float16* __restrict__ pwc0, const _Float16* __restrict__ pwc1,
              const _Float16* __restrict__ pwcond, const _Float16* __restrict__ pwskip,
              const _Float16* __restrict__ pwres,
              const float* __restrict__ causal_b, const float* __restrict__ cond_b,
              const float* __restrict__ skip_b, const float* __restrict__ res_b,
              float* __restrict__ skip, int d, int first) {
  __shared__ _Float16 sRt[64 * 128];   // res[t0 .. t0+64)      time-major
  __shared__ _Float16 sRd[64 * 128];   // res[t0-d .. t0+64-d)  (zeros for t<d)
  __shared__ _Float16 sH [64 * 128];   // gated h
  __shared__ float    zb[256];         // z bias per padded out-channel

  const int tid  = threadIdx.x;
  const int t0   = blockIdx.x * 64;
  const int wave = tid >> 5;
  const int lane = tid & 31;
  const int hi   = lane >> 4;
  const int l15  = lane & 15;

  __builtin_prefetch(pwc1, 0, 1);
  __builtin_prefetch(pwskip, 0, 1);

  // z bias (padded 256 rows: tanh 0..127, sigmoid 128..255)
  {
    int m = tid;
    int om = (m < 128) ? m : (m - 8);
    bool ok = (m < 128) ? (m < 120) : (om < 240);
    zb[m] = ok ? (causal_b[om] + cond_b[om]) : 0.0f;
  }

  // ---- async-stage res tiles (16B chunks; 64 rows x 256B) ----
  #pragma unroll
  for (int it = 0; it < 4; ++it) {
    int e    = it * 256 + tid;
    int row  = e >> 4;            // 0..63
    int ch   = (e & 15) * 8;      // f16 channel offset, 16B chunks
    async_copy16(res_in + (size_t)(t0 + row) * 128 + ch, sRt + row * 128 + ch);
    int td = t0 + row - d;
    if (td >= 0) {
      async_copy16(res_in + (size_t)td * 128 + ch, sRd + row * 128 + ch);
    } else {
      v8h z = {};
      *(v8h*)(sRd + row * 128 + ch) = z;
    }
  }
  async_join();
  __syncthreads();

  // ---- phase 2: z -> gated h ----
  const _Float16* condT = cond + (size_t)t0 * 96;
  for (int jp = wave; jp < 32; jp += 8) {
    int mt = jp & 3;              // time tile 0..3
    int np = jp >> 2;             // tanh N-tile 0..7 (sigmoid = np+8)
    int trow = mt * 16;
    v8f aT = {}; v8f aS = {};
    #pragma unroll
    for (int kt = 0; kt < 4; ++kt) {
      v16h ar = ldAct(sRt, 128, trow, kt, lane);
      v16h ad = ldAct(sRd, 128, trow, kt, lane);
      aT = wmma_f16(ar, ldPW(pwc1, np * 4 + kt, lane), aT);
      aS = wmma_f16(ar, ldPW(pwc1, (np + 8) * 4 + kt, lane), aS);
      aT = wmma_f16(ad, ldPW(pwc0, np * 4 + kt, lane), aT);
      aS = wmma_f16(ad, ldPW(pwc0, (np + 8) * 4 + kt, lane), aS);
    }
    #pragma unroll
    for (int kt = 0; kt < 3; ++kt) {
      v16h ac = ldAct(condT, 96, trow, kt, lane);
      aT = wmma_f16(ac, ldPW(pwcond, np * 3 + kt, lane), aT);
      aS = wmma_f16(ac, ldPW(pwcond, (np + 8) * 3 + kt, lane), aS);
    }
    #pragma unroll
    for (int r = 0; r < 8; ++r) {
      int tloc = trow + r + hi * 8;
      int c    = np * 16 + l15;
      float zT = aT[r] + zb[c];
      float zS = aS[r] + zb[128 + c];
      sH[tloc * 128 + c] = (_Float16)(tanhf(zT) * (1.0f / (1.0f + __expf(-zS))));
    }
  }
  __syncthreads();

  // ---- phase 3: skip (K128 -> N240) and res (K128 -> N128) ----
  for (int jp = wave; jp < 92; jp += 8) {
    if (jp < 60) {
      int mt = jp & 3, nt = jp >> 2;       // nt 0..14
      v8f acc = {};
      #pragma unroll
      for (int kt = 0; kt < 4; ++kt)
        acc = wmma_f16(ldAct(sH, 128, mt * 16, kt, lane),
                       ldPW(pwskip, nt * 4 + kt, lane), acc);
      #pragma unroll
      for (int r = 0; r < 8; ++r) {
        int tloc = mt * 16 + r + hi * 8;
        int oc   = nt * 16 + l15;
        float v = acc[r] + skip_b[oc];
        float* p = skip + (size_t)(t0 + tloc) * 256 + oc;
        *p = first ? v : (*p + v);
      }
    } else {
      int jq = jp - 60;
      int mt = jq & 3, nt = jq >> 2;       // nt 0..7
      v8f acc = {};
      #pragma unroll
      for (int kt = 0; kt < 4; ++kt)
        acc = wmma_f16(ldAct(sH, 128, mt * 16, kt, lane),
                       ldPW(pwres, nt * 4 + kt, lane), acc);
      #pragma unroll
      for (int r = 0; r < 8; ++r) {
        int tloc = mt * 16 + r + hi * 8;
        int c    = nt * 16 + l15;
        float v = acc[r] + ((c < RC) ? res_b[c] : 0.0f)
                + (float)sRt[tloc * 128 + c];
        res_out[(size_t)(t0 + tloc) * 128 + c] = (_Float16)v;
      }
    }
  }
}

// ---------------------------------------------------------------------------
// Head: x = relu(skip); y = relu(x@h1^T + b); out = y@h2^T + b.  32 rows / WG.
// ---------------------------------------------------------------------------
__global__ void __launch_bounds__(256)
wavenet_head(const float* __restrict__ skip,
             const _Float16* __restrict__ pwh1, const _Float16* __restrict__ pwh2,
             const float* __restrict__ h1_b, const float* __restrict__ h2_b,
             float* __restrict__ out, int T) {
  __shared__ _Float16 sX[32 * 256];
  __shared__ _Float16 sY[32 * 256];

  const int tid  = threadIdx.x;
  const int t0   = blockIdx.x * 32;
  const int wave = tid >> 5;
  const int lane = tid & 31;
  const int hi   = lane >> 4;
  const int l15  = lane & 15;

  __builtin_prefetch(pwh1, 0, 1);

  #pragma unroll
  for (int it = 0; it < 32; ++it) {
    int e   = it * 256 + tid;
    int row = e >> 8;
    int c   = e & 255;
    float v = (c < SC) ? fmaxf(skip[(size_t)(t0 + row) * 256 + c], 0.0f) : 0.0f;
    sX[row * 256 + c] = (_Float16)v;
  }
  __syncthreads();

  for (int jp = wave; jp < 32; jp += 8) {
    int mt = jp & 1, nt = jp >> 1;
    v8f acc = {};
    #pragma unroll
    for (int kt = 0; kt < 8; ++kt)
      acc = wmma_f16(ldAct(sX, 256, mt * 16, kt, lane),
                     ldPW(pwh1, nt * 8 + kt, lane), acc);
    #pragma unroll
    for (int r = 0; r < 8; ++r) {
      int tloc = mt * 16 + r + hi * 8;
      int oc   = nt * 16 + l15;
      sY[tloc * 256 + oc] = (_Float16)fmaxf(acc[r] + h1_b[oc], 0.0f);
    }
  }
  __syncthreads();

  for (int jp = wave; jp < 32; jp += 8) {
    int mt = jp & 1, nt = jp >> 1;
    v8f acc = {};
    #pragma unroll
    for (int kt = 0; kt < 8; ++kt)
      acc = wmma_f16(ldAct(sY, 256, mt * 16, kt, lane),
                     ldPW(pwh2, nt * 8 + kt, lane), acc);
    #pragma unroll
    for (int r = 0; r < 8; ++r) {
      int t  = t0 + mt * 16 + r + hi * 8;
      int oc = nt * 16 + l15;
      if (t < T) out[(size_t)oc * T + t] = acc[r] + h2_b[oc];
    }
  }
}

// ---------------------------------------------------------------------------
extern "C" void kernel_launch(void* const* d_in, const int* in_sizes, int n_in,
                              void* d_out, int out_size, void* d_ws, size_t ws_size,
                              hipStream_t stream) {
  const float* wav   = (const float*)d_in[0];
  const float* mel   = (const float*)d_in[1];
  const float* up_w  = (const float*)d_in[2];
  const float* up_b  = (const float*)d_in[3];
  const float* in_w  = (const float*)d_in[4];
  const float* in_b  = (const float*)d_in[5];
  const float* bc_w  = (const float*)d_in[6];
  const float* bc_b  = (const float*)d_in[7];
  const float* bcd_w = (const float*)d_in[8];
  const float* bcd_b = (const float*)d_in[9];
  const float* bs_w  = (const float*)d_in[10];
  const float* bs_b  = (const float*)d_in[11];
  const float* br_w  = (const float*)d_in[12];
  const float* br_b  = (const float*)d_in[13];
  const float* h1_w  = (const float*)d_in[14];
  const float* h1_b  = (const float*)d_in[15];
  const float* h2_w  = (const float*)d_in[16];
  const float* h2_b  = (const float*)d_in[17];
  float* out = (float*)d_out;

  const int T    = out_size / NQ;            // 39904 for the reference shapes
  const int Tpad = ((T + 63) / 64) * 64;

  // ---- workspace carve-up (256B aligned) ----
  char* ws = (char*)d_ws;
  size_t off = 0;
  auto alloc = [&](size_t bytes) {
    char* p = ws + off;
    off = (off + bytes + 255) & ~(size_t)255;
    return p;
  };
  _Float16* cond   = (_Float16*)alloc((size_t)Tpad * 96  * 2);
  _Float16* res_a  = (_Float16*)alloc((size_t)Tpad * 128 * 2);
  _Float16* res_b  = (_Float16*)alloc((size_t)Tpad * 128 * 2);
  float*    skip   = (float*)   alloc((size_t)Tpad * 256 * 4);
  _Float16* pwc0   = (_Float16*)alloc((size_t)NB * 16 * 4 * 512 * 2);
  _Float16* pwc1   = (_Float16*)alloc((size_t)NB * 16 * 4 * 512 * 2);
  _Float16* pwcond = (_Float16*)alloc((size_t)NB * 16 * 3 * 512 * 2);
  _Float16* pwskip = (_Float16*)alloc((size_t)NB * 15 * 4 * 512 * 2);
  _Float16* pwres  = (_Float16*)alloc((size_t)NB * 8  * 4 * 512 * 2);
  _Float16* pwh1   = (_Float16*)alloc((size_t)16 * 8 * 512 * 2);
  _Float16* pwh2   = (_Float16*)alloc((size_t)16 * 8 * 512 * 2);
  (void)ws_size; (void)n_in; (void)in_sizes;

  auto cdiv = [](long a, long b) { return (int)((a + b - 1) / b); };

  // ---- weight repack (consumed as WMMA B operands) ----
  pack_weights<<<cdiv((long)NB*16*4*512,256),256,0,stream>>>(bc_w, pwc0, NB,16,4, 0,120,1, 57600,240,2,0);
  pack_weights<<<cdiv((long)NB*16*4*512,256),256,0,stream>>>(bc_w, pwc1, NB,16,4, 0,120,1, 57600,240,2,1);
  pack_weights<<<cdiv((long)NB*16*3*512,256),256,0,stream>>>(bcd_w, pwcond, NB,16,3, 0,80,1, 19200,80,1,0);
  pack_weights<<<cdiv((long)NB*15*4*512,256),256,0,stream>>>(bs_w, pwskip, NB,15,4, 240,120,0, 28800,120,1,0);
  pack_weights<<<cdiv((long)NB*8*4*512,256),256,0,stream>>>(br_w, pwres, NB,8,4, 120,120,0, 14400,120,1,0);
  pack_weights<<<cdiv((long)16*8*512,256),256,0,stream>>>(h1_w, pwh1, 1,16,8, 256,240,0, 0,240,1,0);
  pack_weights<<<cdiv((long)16*8*512,256),256,0,stream>>>(h2_w, pwh2, 1,16,8, 256,256,0, 0,256,1,0);

  // ---- activations ----
  upsample_cond<<<cdiv((long)96*Tpad,256),256,0,stream>>>(mel, up_w, up_b, cond, Tpad);
  input_conv  <<<cdiv((long)128*Tpad,256),256,0,stream>>>(wav, in_w, in_b, res_a, Tpad);

  // ---- 16 residual blocks ----
  static const int DILS[NB] = {1,2,4,8,16,32,64,128, 1,2,4,8,16,32,64,128};
  const int nwg = Tpad / 64;
  for (int i = 0; i < NB; ++i) {
    const _Float16* rin  = (i & 1) ? res_b : res_a;
    _Float16*       rout = (i & 1) ? res_a : res_b;
    wavenet_block<<<nwg,256,0,stream>>>(
        rin, rout, cond,
        pwc0   + (size_t)i * 16 * 4 * 512,
        pwc1   + (size_t)i * 16 * 4 * 512,
        pwcond + (size_t)i * 16 * 3 * 512,
        pwskip + (size_t)i * 15 * 4 * 512,
        pwres  + (size_t)i * 8  * 4 * 512,
        bc_b + i * SC, bcd_b + i * SC, bs_b + i * SC, br_b + i * RC,
        skip, DILS[i], (i == 0) ? 1 : 0);
  }

  // ---- head ----
  wavenet_head<<<cdiv(T,32),256,0,stream>>>(skip, pwh1, pwh2, h1_b, h2_b, out, T);
}
